// E3EquivariantGNN_73993696575533
// MI455X (gfx1250) — matrix operand order repaired and checked by reading, not compiled, not run on hardware
//
#include <hip/hip_runtime.h>
#include <hip/hip_bf16.h>

#define HID     128
#define NNODES  50000
#define NEDGES  500000
#define NLAYER  4
#define NGRAPH  64
#define NODEF   15

// dynamic LDS: packed W1 (256x128 bf16) + packed W2 (128x128 bf16) + 8-wave stage
#define SMEM_BYTES ((256 * HID + HID * HID + 8 * 16 * HID) * 2)

typedef __attribute__((ext_vector_type(16))) __bf16 v16bf;
typedef __attribute__((ext_vector_type(8)))  __bf16 bf16x8;
typedef __attribute__((ext_vector_type(8)))  float  v8f;

__device__ __forceinline__ float frelu(float x) { return x > 0.f ? x : 0.f; }

// ---- WMMA helpers -----------------------------------------------------------
__device__ __forceinline__ v8f wmma_bf16(v16bf a, v16bf b, v8f c) {
    // D = A(16x32 bf16) * B(32x16 bf16) + C(16x16 f32)
    return __builtin_amdgcn_wmma_f32_16x16x32_bf16(
        /*neg_a=*/false, a, /*neg_b=*/false, b,
        /*c_mod=*/(short)0, c, /*reuse_a=*/false, /*reuse_b=*/false);
}

// A fragment: lane l holds row (l&15); 16 bf16 = K {c0..c0+7, c0+16..c0+23}
// where c0 already includes the 8*(l>>4) half-offset. Two 16B contiguous loads.
__device__ __forceinline__ v16bf load_a_bf16(const __bf16* row, int c0) {
    v16bf a;
    ((bf16x8*)&a)[0] = *(const bf16x8*)(row + c0);
    ((bf16x8*)&a)[1] = *(const bf16x8*)(row + c0 + 16);
    return a;
}
__device__ __forceinline__ v16bf load_a_f32(const float* row, int c0) {
    bf16x8 t0, t1;
#pragma unroll
    for (int i = 0; i < 8; i++) {
        t0[i] = (__bf16)row[c0 + i];
        t1[i] = (__bf16)row[c0 + 16 + i];
    }
    v16bf a;
    ((bf16x8*)&a)[0] = t0;
    ((bf16x8*)&a)[1] = t1;
    return a;
}
// B fragment from LDS: 32 contiguous bytes per lane.
__device__ __forceinline__ v16bf load_b_lds(const __bf16* base, int tile, int lane) {
    return *(const v16bf*)(base + tile * 512 + lane * 16);
}

// ---- Weight packing: fp32 [L,K,N] row-major -> bf16 B-tile layout -----------
// Tile (kt,nt) = 512 bf16 contiguous: lane l (<16): (K=kt*32+{i<8: i, else 8+i},     N=nt*16+l)
//                                     lane l (>=16):(K=kt*32+{i<8: 8+i, else 16+i}, N=nt*16+l-16)
__global__ __launch_bounds__(256)
void gnn_pack_b(const float* __restrict__ src, __bf16* __restrict__ dst,
                int L, int K, int N, int srcLayerStride) {
    int idx = blockIdx.x * 256 + threadIdx.x;
    int per = K * N;
    if (idx >= L * per) return;
    int layer = idx / per, rem = idx - layer * per;
    int k = rem / N, n = rem - k * N;
    int kt = k >> 5, kr = k & 31, nt = n >> 4, nr = n & 15;
    int lane = ((kr >> 3) & 1) ? (16 + nr) : nr;
    int i = (kr & 7) + ((kr >> 4) << 3);
    size_t o = (size_t)layer * per + ((size_t)(kt * (N >> 4) + nt)) * 512 + lane * 16 + i;
    dst[o] = (__bf16)src[(size_t)layer * srcLayerStride + (size_t)k * N + n];
}

// ---- Node encoder: h = x @ enc_w + enc_b (fp32 master + bf16 mirror) --------
__global__ __launch_bounds__(256)
void gnn_encode(const float* __restrict__ x, const float* __restrict__ w,
                const float* __restrict__ b, float* __restrict__ h32,
                __bf16* __restrict__ h16) {
    int i = blockIdx.x * 256 + threadIdx.x;
    if (i >= NNODES * HID) return;
    int node = i >> 7, c = i & 127;
    const float* xr = x + node * NODEF;
    float s = b[c];
#pragma unroll
    for (int k = 0; k < NODEF; k++) s += xr[k] * w[k * HID + c];
    h32[i] = s;
    h16[i] = (__bf16)s;
}

// ---- Edge distances ---------------------------------------------------------
__global__ __launch_bounds__(256)
void gnn_dist(const float* __restrict__ pos, const long long* __restrict__ srcI,
              const long long* __restrict__ dstI, float* __restrict__ dist) {
    int e = blockIdx.x * 256 + threadIdx.x;
    if (e >= NEDGES) return;
    long long s = srcI[e], d = dstI[e];
    float dx = pos[d * 3 + 0] - pos[s * 3 + 0];
    float dy = pos[d * 3 + 1] - pos[s * 3 + 1];
    float dz = pos[d * 3 + 2] - pos[s * 3 + 2];
    dist[e] = sqrtf(dx * dx + dy * dy + dz * dz);
}

// ---- Edge message MLP + scatter-add ----------------------------------------
// Block = 8 waves. Packed weights staged into LDS once per block (shared),
// then one wave per 16-edge tile: Stage1 [16,256]x[256,128] bf16 WMMA,
// epilogue bias + dist rank-1 + relu -> LDS transpose; Stage2
// [16,128]x[128,128] WMMA; f32 atomic scatter into aggr[dst].
__global__ __launch_bounds__(256)
void gnn_edge_msg(const __bf16* __restrict__ h16, const long long* __restrict__ srcI,
                  const long long* __restrict__ dstI, const float* __restrict__ dist,
                  const __bf16* __restrict__ w1p, const float* __restrict__ w1last,
                  const float* __restrict__ b1, const __bf16* __restrict__ w2p,
                  const float* __restrict__ b2, float* __restrict__ aggr, int ntiles) {
    extern __shared__ __align__(16) char smem_raw[];
    __bf16* sw1   = (__bf16*)smem_raw;        // 256x128 packed tiles (64KB)
    __bf16* sw2   = sw1 + 256 * HID;          // 128x128 packed tiles (32KB)
    __bf16* stage = sw2 + HID * HID;          // 8 waves x 16 x 128   (32KB)

    // Cooperative weight staging: global (L2-resident) -> LDS, 16B vectors.
    {
        const uint4* g1 = (const uint4*)w1p;
        const uint4* g2 = (const uint4*)w2p;
        uint4* l1 = (uint4*)sw1;
        uint4* l2 = (uint4*)sw2;
        for (int i = threadIdx.x; i < (256 * HID) / 8; i += 256) l1[i] = g1[i];
        for (int i = threadIdx.x; i < (HID * HID) / 8; i += 256) l2[i] = g2[i];
    }
    __syncthreads();

    const int wave = threadIdx.x >> 5;
    const int lane = threadIdx.x & 31;
    const int tile = blockIdx.x * 8 + wave;
    if (tile >= ntiles) return;
    const int ebase = tile * 16;
    const int r    = lane & 15;        // A-row / C-column index for this lane
    const int kh   = (lane >> 4) * 8;  // K half-offset
    const int mrow = (lane >> 4) * 8;  // C-row base for this lane

    const long long nj = srcI[ebase + r];   // x_j = h[src]
    const long long ni = dstI[ebase + r];   // x_i = h[dst]
    const __bf16* rowi = h16 + (size_t)ni * HID;
    const __bf16* rowj = h16 + (size_t)nj * HID;

    v8f acc[8];
#pragma unroll
    for (int nt = 0; nt < 8; nt++)
#pragma unroll
        for (int j = 0; j < 8; j++) acc[nt][j] = 0.f;

    // K = 0..255: 4 k-tiles from x_i, 4 from x_j
#pragma unroll
    for (int kt = 0; kt < 8; kt++) {
        const __bf16* rp = (kt < 4) ? rowi : rowj;
        v16bf a = load_a_bf16(rp, (kt & 3) * 32 + kh);
#pragma unroll
        for (int nt = 0; nt < 8; nt++)
            acc[nt] = wmma_bf16(a, load_b_lds(sw1, kt * 8 + nt, lane), acc[nt]);
    }
    // Epilogue: + bias + dist column (row 256 of msg_w1), relu -> LDS stage.
    float dv[8];
#pragma unroll
    for (int j = 0; j < 8; j++) dv[j] = dist[ebase + mrow + j];
    __bf16* mst = stage + wave * 16 * HID;
#pragma unroll
    for (int nt = 0; nt < 8; nt++) {
        float wl = w1last[nt * 16 + r];
        float bv = b1[nt * 16 + r];
#pragma unroll
        for (int j = 0; j < 8; j++) {
            float v = frelu(acc[nt][j] + bv + dv[j] * wl);
            mst[(mrow + j) * HID + nt * 16 + r] = (__bf16)v;
        }
    }
    // Stage 2: m @ msg_w2
    v8f acc2[8];
#pragma unroll
    for (int nt = 0; nt < 8; nt++)
#pragma unroll
        for (int j = 0; j < 8; j++) acc2[nt][j] = 0.f;
#pragma unroll
    for (int kt = 0; kt < 4; kt++) {
        v16bf a = load_a_bf16(mst + r * HID, kt * 32 + kh);
#pragma unroll
        for (int nt = 0; nt < 8; nt++)
            acc2[nt] = wmma_bf16(a, load_b_lds(sw2, kt * 8 + nt, lane), acc2[nt]);
    }
    // Scatter-add into aggr[dst] (bias folded in here).
    long long drow[8];
#pragma unroll
    for (int j = 0; j < 8; j++) drow[j] = dstI[ebase + mrow + j];
#pragma unroll
    for (int nt = 0; nt < 8; nt++) {
        float bv = b2[nt * 16 + r];
#pragma unroll
        for (int j = 0; j < 8; j++)
            atomicAdd(&aggr[(size_t)drow[j] * HID + nt * 16 + r], acc2[nt][j] + bv);
    }
}

// ---- Node update MLP + LayerNorm + relu + residual --------------------------
__global__ __launch_bounds__(256)
void gnn_node_update(__bf16* h16, const float* __restrict__ aggr,
                     const __bf16* __restrict__ w1p, const float* __restrict__ b1,
                     const __bf16* __restrict__ w2p, const float* __restrict__ b2,
                     const float* __restrict__ lng, const float* __restrict__ lnb,
                     float* h32, int ntiles, int isfirst) {
    extern __shared__ __align__(16) char smem_raw[];
    __bf16* sw1   = (__bf16*)smem_raw;
    __bf16* sw2   = sw1 + 256 * HID;
    __bf16* stage = sw2 + HID * HID;
    {
        const uint4* g1 = (const uint4*)w1p;
        const uint4* g2 = (const uint4*)w2p;
        uint4* l1 = (uint4*)sw1;
        uint4* l2 = (uint4*)sw2;
        for (int i = threadIdx.x; i < (256 * HID) / 8; i += 256) l1[i] = g1[i];
        for (int i = threadIdx.x; i < (HID * HID) / 8; i += 256) l2[i] = g2[i];
    }
    __syncthreads();

    const int wave = threadIdx.x >> 5;
    const int lane = threadIdx.x & 31;
    const int tile = blockIdx.x * 8 + wave;
    if (tile >= ntiles) return;
    const int nbase = tile * 16;
    const int r    = lane & 15;
    const int kh   = (lane >> 4) * 8;
    const int mrow = (lane >> 4) * 8;

    const __bf16* hrow = h16 + (size_t)(nbase + r) * HID;
    const float*  arow = aggr + (size_t)(nbase + r) * HID;

    v8f acc[8];
#pragma unroll
    for (int nt = 0; nt < 8; nt++)
#pragma unroll
        for (int j = 0; j < 8; j++) acc[nt][j] = 0.f;
    // u_in = [h | aggr] : K 0..127 from bf16 h, K 128..255 from f32 aggr
#pragma unroll
    for (int kt = 0; kt < 8; kt++) {
        v16bf a = (kt < 4) ? load_a_bf16(hrow, kt * 32 + kh)
                           : load_a_f32(arow, (kt - 4) * 32 + kh);
#pragma unroll
        for (int nt = 0; nt < 8; nt++)
            acc[nt] = wmma_bf16(a, load_b_lds(sw1, kt * 8 + nt, lane), acc[nt]);
    }
    __bf16* mst = stage + wave * 16 * HID;
#pragma unroll
    for (int nt = 0; nt < 8; nt++) {
        float bv = b1[nt * 16 + r];
#pragma unroll
        for (int j = 0; j < 8; j++)
            mst[(mrow + j) * HID + nt * 16 + r] = (__bf16)frelu(acc[nt][j] + bv);
    }
    v8f acc2[8];
#pragma unroll
    for (int nt = 0; nt < 8; nt++)
#pragma unroll
        for (int j = 0; j < 8; j++) acc2[nt][j] = 0.f;
#pragma unroll
    for (int kt = 0; kt < 4; kt++) {
        v16bf a = load_a_bf16(mst + r * HID, kt * 32 + kh);
#pragma unroll
        for (int nt = 0; nt < 8; nt++)
            acc2[nt] = wmma_bf16(a, load_b_lds(sw2, kt * 8 + nt, lane), acc2[nt]);
    }
    // Fold bias, then LayerNorm across the 128 outputs of each row:
    // per-lane partials, reduce across the 16 lanes of each half (shfl_xor).
#pragma unroll
    for (int nt = 0; nt < 8; nt++) {
        float bv = b2[nt * 16 + r];
#pragma unroll
        for (int j = 0; j < 8; j++) acc2[nt][j] += bv;
    }
    float s[8], q[8];
#pragma unroll
    for (int j = 0; j < 8; j++) { s[j] = 0.f; q[j] = 0.f; }
#pragma unroll
    for (int nt = 0; nt < 8; nt++)
#pragma unroll
        for (int j = 0; j < 8; j++) {
            float v = acc2[nt][j];
            s[j] += v; q[j] += v * v;
        }
#pragma unroll
    for (int m = 1; m < 16; m <<= 1) {
#pragma unroll
        for (int j = 0; j < 8; j++) {
            s[j] += __shfl_xor(s[j], m, 32);
            q[j] += __shfl_xor(q[j], m, 32);
        }
    }
    float mu[8], rs[8];
#pragma unroll
    for (int j = 0; j < 8; j++) {
        mu[j] = s[j] * (1.f / HID);
        float var = q[j] * (1.f / HID) - mu[j] * mu[j];
        rs[j] = rsqrtf(var + 1e-5f);
    }
#pragma unroll
    for (int nt = 0; nt < 8; nt++) {
        float gv = lng[nt * 16 + r];
        float bv = lnb[nt * 16 + r];
#pragma unroll
        for (int j = 0; j < 8; j++) {
            float u = frelu((acc2[nt][j] - mu[j]) * rs[j] * gv + bv);
            int idx = (nbase + mrow + j) * HID + nt * 16 + r;
            float hn = isfirst ? u : (h32[idx] + u);
            h32[idx] = hn;
            h16[idx] = (__bf16)hn;
        }
    }
}

// ---- Global mean pool (scatter) ---------------------------------------------
__global__ __launch_bounds__(256)
void gnn_pool(const float* __restrict__ h32, const long long* __restrict__ batch,
              float* __restrict__ pooled, float* __restrict__ cnt) {
    int i = blockIdx.x * 256 + threadIdx.x;
    if (i >= NNODES * HID) return;
    int node = i >> 7, c = i & 127;
    long long g = batch[node];
    atomicAdd(&pooled[(size_t)g * HID + c], h32[i]);
    if (c == 0) atomicAdd(&cnt[g], 1.0f);
}

// ---- Readout MLP (64 graphs, tiny) ------------------------------------------
__global__ __launch_bounds__(128)
void gnn_readout(const float* __restrict__ pooled, const float* __restrict__ cnt,
                 const float* __restrict__ w1, const float* __restrict__ b1,
                 const float* __restrict__ w2, const float* __restrict__ b2,
                 const float* __restrict__ w3, const float* __restrict__ b3,
                 float* __restrict__ out) {
    __shared__ float p[HID], s1[HID], s2[HID / 2];
    int g = blockIdx.x, t = threadIdx.x;
    float c = cnt[g];
    c = c > 1.f ? c : 1.f;
    p[t] = pooled[g * HID + t] / c;
    __syncthreads();
    float a = b1[t];
    for (int k = 0; k < HID; k++) a += p[k] * w1[k * HID + t];
    s1[t] = frelu(a);
    __syncthreads();
    if (t < HID / 2) {
        float a2 = b2[t];
        for (int k = 0; k < HID; k++) a2 += s1[k] * w2[k * (HID / 2) + t];
        s2[t] = frelu(a2);
    }
    __syncthreads();
    if (t == 0) {
        float o = b3[0];
        for (int k = 0; k < HID / 2; k++) o += s2[k] * w3[k];
        out[g] = o;
    }
}

// ---- Host launch ------------------------------------------------------------
extern "C" void kernel_launch(void* const* d_in, const int* in_sizes, int n_in,
                              void* d_out, int out_size, void* d_ws, size_t ws_size,
                              hipStream_t stream) {
    (void)in_sizes; (void)n_in; (void)out_size; (void)ws_size;
    const float*     x      = (const float*)d_in[0];
    const long long* ei     = (const long long*)d_in[1];
    // d_in[2] edge_attr: encoded but unused downstream in the reference -> skipped
    const float*     pos    = (const float*)d_in[3];
    const long long* batch  = (const long long*)d_in[4];
    const float*     enc_w  = (const float*)d_in[5];
    const float*     enc_b  = (const float*)d_in[6];
    const float*     msg_w1 = (const float*)d_in[9];
    const float*     msg_b1 = (const float*)d_in[10];
    const float*     msg_w2 = (const float*)d_in[11];
    const float*     msg_b2 = (const float*)d_in[12];
    const float*     upd_w1 = (const float*)d_in[13];
    const float*     upd_b1 = (const float*)d_in[14];
    const float*     upd_w2 = (const float*)d_in[15];
    const float*     upd_b2 = (const float*)d_in[16];
    const float*     ln_g   = (const float*)d_in[17];
    const float*     ln_b   = (const float*)d_in[18];
    const float*     mlp_w1 = (const float*)d_in[19];
    const float*     mlp_b1 = (const float*)d_in[20];
    const float*     mlp_w2 = (const float*)d_in[21];
    const float*     mlp_b2 = (const float*)d_in[22];
    const float*     mlp_w3 = (const float*)d_in[23];
    const float*     mlp_b3 = (const float*)d_in[24];
    float* out = (float*)d_out;

    char* wsp = (char*)d_ws;
    auto carve = [&](size_t bytes) -> char* {
        char* p = wsp;
        wsp += (bytes + 255) & ~(size_t)255;
        return p;
    };
    float*  h32    = (float*)carve((size_t)NNODES * HID * 4);
    __bf16* h16    = (__bf16*)carve((size_t)NNODES * HID * 2);
    float*  aggr   = (float*)carve((size_t)NNODES * HID * 4);
    float*  dist   = (float*)carve((size_t)NEDGES * 4);
    __bf16* w1p    = (__bf16*)carve((size_t)NLAYER * 256 * HID * 2);
    __bf16* w2p    = (__bf16*)carve((size_t)NLAYER * HID * HID * 2);
    __bf16* uw1p   = (__bf16*)carve((size_t)NLAYER * 256 * HID * 2);
    __bf16* uw2p   = (__bf16*)carve((size_t)NLAYER * HID * HID * 2);
    float*  pooled = (float*)carve((size_t)NGRAPH * HID * 4);
    float*  cnt    = (float*)carve((size_t)NGRAPH * 4);

    const long long* srcI = ei;            // edge_index[0]
    const long long* dstI = ei + NEDGES;   // edge_index[1]

    // Pack weights into WMMA B-tile bf16 layout (once per launch).
    {
        int t1 = NLAYER * 256 * HID, t2 = NLAYER * HID * HID;
        gnn_pack_b<<<(t1 + 255) / 256, 256, 0, stream>>>(msg_w1, w1p, NLAYER, 256, HID, 257 * HID);
        gnn_pack_b<<<(t2 + 255) / 256, 256, 0, stream>>>(msg_w2, w2p, NLAYER, HID, HID, HID * HID);
        gnn_pack_b<<<(t1 + 255) / 256, 256, 0, stream>>>(upd_w1, uw1p, NLAYER, 256, HID, 256 * HID);
        gnn_pack_b<<<(t2 + 255) / 256, 256, 0, stream>>>(upd_w2, uw2p, NLAYER, HID, HID, HID * HID);
    }
    gnn_encode<<<(NNODES * HID + 255) / 256, 256, 0, stream>>>(x, enc_w, enc_b, h32, h16);
    gnn_dist<<<(NEDGES + 255) / 256, 256, 0, stream>>>(pos, srcI, dstI, dist);

    const int etiles = NEDGES / 16;   // 31250 (exact)
    const int vtiles = NNODES / 16;   // 3125  (exact)
    for (int l = 0; l < NLAYER; l++) {
        hipMemsetAsync(aggr, 0, (size_t)NNODES * HID * 4, stream);
        gnn_edge_msg<<<(etiles + 7) / 8, 256, SMEM_BYTES, stream>>>(
            h16, srcI, dstI, dist,
            w1p + (size_t)l * 256 * HID,
            msg_w1 + (size_t)l * 257 * HID + 256 * HID,  // dist row of msg_w1
            msg_b1 + l * HID,
            w2p + (size_t)l * HID * HID,
            msg_b2 + l * HID, aggr, etiles);
        gnn_node_update<<<(vtiles + 7) / 8, 256, SMEM_BYTES, stream>>>(
            h16, aggr,
            uw1p + (size_t)l * 256 * HID, upd_b1 + l * HID,
            uw2p + (size_t)l * HID * HID, upd_b2 + l * HID,
            ln_g + l * HID, ln_b + l * HID, h32, vtiles, l == 0 ? 1 : 0);
    }
    hipMemsetAsync(pooled, 0, (size_t)NGRAPH * HID * 4, stream);
    hipMemsetAsync(cnt, 0, (size_t)NGRAPH * 4, stream);
    gnn_pool<<<(NNODES * HID + 255) / 256, 256, 0, stream>>>(h32, batch, pooled, cnt);
    gnn_readout<<<NGRAPH, HID, 0, stream>>>(pooled, cnt, mlp_w1, mlp_b1, mlp_w2, mlp_b2,
                                            mlp_w3, mlp_b3, out);
}
